// UniCrystalFormer_16123307229968
// MI455X (gfx1250) — matrix operand order repaired and checked by reading, not compiled, not run on hardware
//
#include <hip/hip_runtime.h>

// ---------------------------------------------------------------------------
// UniCrystalFormer forward for MI455X (gfx1250, wave32, WMMA f16 16x16x32)
// All GEMM operands are padded to 64-row multiples so the WMMA GEMM has no
// bounds checks: unconditional b128 global loads -> LDS -> b128 fragment
// loads -> v_wmma_f32_16x16x32_f16.
// ---------------------------------------------------------------------------

typedef __attribute__((ext_vector_type(16))) _Float16 v16h;
typedef __attribute__((ext_vector_type(8)))  _Float16 v8h;
typedef __attribute__((ext_vector_type(8)))  float    v8f;

static constexpr int NN  = 10000;   // nodes
static constexpr int NNP = 10048;   // nodes padded to 64
static constexpr int NE  = 32000;   // edges (already 64-multiple)
static constexpr int NG  = 200;     // graphs
static constexpr int NGP = 256;     // graphs padded to 64
static constexpr int HD  = 128;     // hidden

__device__ __forceinline__ float sigf(float x) { return 1.f / (1.f + __expf(-x)); }
__device__ __forceinline__ float siluf(float x) { return x * sigf(x); }

// ---------------------------------------------------------------------------
// Weight conversion: W [K,N] f32 row-major -> Wt [N,Kpad] f16 (zero pad K..Kpad)
// ---------------------------------------------------------------------------
__global__ void k_transpose_cvt(const float* __restrict__ W, _Float16* __restrict__ Wt,
                                int K, int Kpad, int N) {
  int t = blockIdx.x * blockDim.x + threadIdx.x;
  if (t >= N * Kpad) return;
  int n = t / Kpad, k = t - n * Kpad;
  Wt[t] = (k < K) ? (_Float16)W[(size_t)k * N + n] : (_Float16)0.f;
}

// ---------------------------------------------------------------------------
// WMMA GEMM: C[M,N] = act(A[M,K] @ B[K,N] + bias), B transposed [N,K] f16.
// 128 threads = 4 waves; block tile 64x64; K step 32; K%32==0; rows padded.
// No conditionals anywhere: EXEC stays all-ones (WMMA requirement) and the
// staging path is pure b128 traffic.
// ---------------------------------------------------------------------------
__global__ __launch_bounds__(128)
void k_gemm_wmma(const _Float16* __restrict__ A, const _Float16* __restrict__ Bt,
                 const float* __restrict__ bias, float* __restrict__ C,
                 _Float16* __restrict__ C16, int N, int K, int act) {
  __shared__ _Float16 As[64][48];   // 96B row stride keeps 16B alignment
  __shared__ _Float16 Bs[64][48];   // Bs[n][k]
  const int bm = blockIdx.x * 64;
  const int bn = blockIdx.y * 64;
  const int tid = threadIdx.x;
  const int wave = tid >> 5;
  const int lane = tid & 31;

  v8f acc[4];
#pragma unroll
  for (int t = 0; t < 4; ++t)
#pragma unroll
    for (int i = 0; i < 8; ++i) acc[t][i] = 0.f;

  const int lr = tid >> 1;          // 0..63 cooperative load row
  const int lc = (tid & 1) * 16;    // 0 or 16 k-offset
  const _Float16* aPtr = A + (size_t)(bm + lr) * K + lc;
  const _Float16* bPtr = Bt + (size_t)(bn + lr) * K + lc;

  for (int k0 = 0; k0 < K; k0 += 32) {
    v8h a0 = *(const v8h*)(aPtr);
    v8h a1 = *(const v8h*)(aPtr + 8);
    v8h b0 = *(const v8h*)(bPtr);
    v8h b1 = *(const v8h*)(bPtr + 8);
    __builtin_prefetch((const void*)(aPtr + 32), 0, 0);
    __builtin_prefetch((const void*)(bPtr + 32), 0, 0);
    *(v8h*)&As[lr][lc]     = a0;
    *(v8h*)&As[lr][lc + 8] = a1;
    *(v8h*)&Bs[lr][lc]     = b0;
    *(v8h*)&Bs[lr][lc + 8] = b1;
    aPtr += 32; bPtr += 32;
    __syncthreads();

    // A fragment (16-bit A 16x32, k order matched to B):
    // lane<16: K0..15 ; lane>=16: K16..31
    v16h af;
    {
      const int r = lane & 15;
      const int g = (lane >> 4) * 16;
      const _Float16* ap = &As[wave * 16 + r][0];
#pragma unroll
      for (int j = 0; j < 8; ++j) af[j] = ap[g + j];
#pragma unroll
      for (int j = 0; j < 8; ++j) af[8 + j] = ap[g + 8 + j];
    }
#pragma unroll
    for (int t = 0; t < 4; ++t) {
      // B fragment: col = lane&15, lanes<16 hold K0..15, lanes>=16 K16..31
      v16h bf;
      const _Float16* bp = &Bs[t * 16 + (lane & 15)][(lane >> 4) * 16];
#pragma unroll
      for (int j = 0; j < 16; ++j) bf[j] = bp[j];
      acc[t] = __builtin_amdgcn_wmma_f32_16x16x32_f16(false, af, false, bf,
                                                      (short)0, acc[t], false, false);
    }
    __syncthreads();
  }

  // epilogue (C/D layout: VGPR i -> row i + 8*(lane>=16), col = lane&15)
  const int rbase = (lane >> 4) * 8;
  const int c16 = lane & 15;
#pragma unroll
  for (int t = 0; t < 4; ++t) {
    const int n = bn + t * 16 + c16;
    const float bv = bias ? bias[n] : 0.f;
#pragma unroll
    for (int i = 0; i < 8; ++i) {
      const int m = bm + wave * 16 + rbase + i;
      float x = acc[t][i] + bv;
      if (act == 1) x = x * (1.f / (1.f + __expf(-x)));        // silu
      else if (act == 2) x = 1.f / (1.f + __expf(-x));          // sigmoid
      if (C)   C[(size_t)m * N + n] = x;
      if (C16) C16[(size_t)m * N + n] = (_Float16)x;
    }
  }
}

// ---------------------------------------------------------------------------
// LayerNorm over 128 features, one wave per row. act: 0 none, 1 sigmoid
// ---------------------------------------------------------------------------
__global__ __launch_bounds__(128)
void k_layernorm128(const float* __restrict__ x, const float* __restrict__ w,
                    const float* __restrict__ b, float* __restrict__ y,
                    _Float16* __restrict__ y16, int rows, int act) {
  int row = blockIdx.x * 4 + (threadIdx.x >> 5);
  if (row >= rows) return;
  int lane = threadIdx.x & 31;
  const float* xr = x + (size_t)row * HD;
  float v[4];
  float s = 0.f;
  for (int j = 0; j < 4; ++j) { v[j] = xr[lane + 32 * j]; s += v[j]; }
  for (int off = 16; off; off >>= 1) s += __shfl_xor(s, off, 32);
  float mean = s * (1.f / 128.f);
  float vs = 0.f;
  for (int j = 0; j < 4; ++j) { float d = v[j] - mean; vs += d * d; }
  for (int off = 16; off; off >>= 1) vs += __shfl_xor(vs, off, 32);
  float inv = rsqrtf(vs * (1.f / 128.f) + 1e-5f);
  for (int j = 0; j < 4; ++j) {
    int c = lane + 32 * j;
    float o = (v[j] - mean) * inv * w[c] + b[c];
    if (act == 1) o = sigf(o);
    if (y) y[(size_t)row * HD + c] = o;
    if (y16) y16[(size_t)row * HD + c] = (_Float16)o;
  }
}

// ---------------------------------------------------------------------------
// Elementwise / gather / scatter kernels
// ---------------------------------------------------------------------------
__global__ void k_embed(const int* __restrict__ an, const float* __restrict__ tab,
                        float* __restrict__ out, int n) {
  int t = blockIdx.x * blockDim.x + threadIdx.x;
  if (t >= n * HD) return;
  int i = t >> 7, j = t & 127;
  out[t] = tab[(size_t)an[i] * HD + j];
}

__global__ void k_pack(const float* __restrict__ x, _Float16* __restrict__ y, int n) {
  int t = blockIdx.x * blockDim.x + threadIdx.x;
  if (t < n) y[t] = (_Float16)x[t];
}

// megnet [rows,16] f32 -> [rows,32] f16 with zero pad (K padded for WMMA)
__global__ void k_pack_meg(const float* __restrict__ x, _Float16* __restrict__ y, int rows) {
  int t = blockIdx.x * blockDim.x + threadIdx.x;
  if (t >= rows * 32) return;
  int i = t >> 5, j = t & 31;
  y[t] = (j < 16) ? (_Float16)x[(size_t)i * 16 + j] : (_Float16)0.f;
}

__global__ void k_cat2(const float* __restrict__ a, const float* __restrict__ b,
                       _Float16* __restrict__ o, int rows) {
  int t = blockIdx.x * blockDim.x + threadIdx.x;
  if (t >= rows * 256) return;
  int i = t >> 8, j = t & 255;
  float v = (j < HD) ? a[(size_t)i * HD + j] : b[(size_t)i * HD + (j - HD)];
  o[t] = (_Float16)v;
}

__global__ void k_fuse(const float* __restrict__ g, const float* __restrict__ emb,
                       const float* __restrict__ mg, _Float16* __restrict__ out, int n) {
  int t = blockIdx.x * blockDim.x + threadIdx.x;
  if (t >= n) return;
  float gv = g[t];
  out[t] = (_Float16)(gv * emb[t] + (1.f - gv) * mg[t]);
}

__global__ void k_rbf(const float* __restrict__ ev, _Float16* __restrict__ rbf, int E) {
  int t = blockIdx.x * blockDim.x + threadIdx.x;
  if (t >= E * HD) return;
  int e = t >> 7, j = t & 127;
  float x0 = ev[e * 3], x1 = ev[e * 3 + 1], x2 = ev[e * 3 + 2];
  float d = sqrtf(x0 * x0 + x1 * x1 + x2 * x2);
  float c = (8.0f / 127.0f) * (float)j;
  float u = d - c;
  rbf[t] = (_Float16)__expf(-(127.0f / 8.0f) * u * u);
}

// cat = [node[dst], node[src], edge]  -> f16 [E,384]
__global__ void k_gather_cat3(const float* __restrict__ nf, const float* __restrict__ ef,
                              const int* __restrict__ dst, const int* __restrict__ src,
                              _Float16* __restrict__ cat, int E) {
  int t = blockIdx.x * blockDim.x + threadIdx.x;
  if (t >= E * HD) return;
  int e = t >> 7, j = t & 127;
  int d = dst[e], s = src[e];
  _Float16* c = cat + (size_t)e * 384;
  c[j]          = (_Float16)nf[(size_t)d * HD + j];
  c[HD + j]     = (_Float16)nf[(size_t)s * HD + j];
  c[2 * HD + j] = (_Float16)ef[(size_t)e * HD + j];
}

__global__ void k_gate_mul(const float* __restrict__ enew, const float* __restrict__ w,
                           const float* __restrict__ b, const float* __restrict__ aggr,
                           float* __restrict__ out, int E) {
  int t = blockIdx.x * blockDim.x + threadIdx.x;
  if (t >= E * HD) return;
  int j = t & 127;
  out[t] = sigf(enew[t] * w[j] + b[j]) * aggr[t];
}

__global__ void k_scatter_add(const float* __restrict__ vals, const int* __restrict__ idx,
                              float* __restrict__ out, int rows) {
  int t = blockIdx.x * blockDim.x + threadIdx.x;
  if (t >= rows * HD) return;
  int e = t >> 7, j = t & 127;
  atomicAdd(&out[(size_t)idx[e] * HD + j], vals[t]);
}

__global__ void k_silu_affine_add(const float* __restrict__ agg, const float* __restrict__ w,
                                  const float* __restrict__ b, const float* __restrict__ xprev,
                                  float* __restrict__ out, int n) {
  int t = blockIdx.x * blockDim.x + threadIdx.x;
  if (t >= n * HD) return;
  int j = t & 127;
  float a = agg[t] * w[j] + b[j];
  out[t] = siluf(a) + xprev[t];
}

__global__ void k_add(const float* __restrict__ a, const float* __restrict__ b,
                      float* __restrict__ c, int n) {
  int t = blockIdx.x * blockDim.x + threadIdx.x;
  if (t < n) c[t] = a[t] + b[t];
}

__global__ void k_count(const int* __restrict__ batch, int* __restrict__ cnt, int n) {
  int i = blockIdx.x * blockDim.x + threadIdx.x;
  if (i < n) atomicAdd(&cnt[batch[i]], 1);
}

__global__ void k_seg_accum(const float* __restrict__ x, const int* __restrict__ batch,
                            float* __restrict__ out, int n) {
  int t = blockIdx.x * blockDim.x + threadIdx.x;
  if (t >= n * HD) return;
  int i = t >> 7, j = t & 127;
  atomicAdd(&out[(size_t)batch[i] * HD + j], x[t]);
}

__global__ void k_gn1(const float* __restrict__ x, const float* __restrict__ gsum,
                      const int* __restrict__ cnt, const float* __restrict__ ms,
                      const int* __restrict__ batch, float* __restrict__ out,
                      float* __restrict__ vsum, int n) {
  int t = blockIdx.x * blockDim.x + threadIdx.x;
  if (t >= n * HD) return;
  int i = t >> 7, j = t & 127;
  int b = batch[i];
  float c = fmaxf((float)cnt[b], 1.f);
  float o = x[t] - ms[j] * (gsum[(size_t)b * HD + j] / c);
  out[t] = o;
  atomicAdd(&vsum[(size_t)b * HD + j], o * o);
}

__global__ void k_gn2(const float* __restrict__ x, const float* __restrict__ vsum,
                      const int* __restrict__ cnt, const int* __restrict__ batch,
                      const float* __restrict__ w, const float* __restrict__ b,
                      float* __restrict__ y, int n) {
  int t = blockIdx.x * blockDim.x + threadIdx.x;
  if (t >= n * HD) return;
  int i = t >> 7, j = t & 127;
  int g = batch[i];
  float c = fmaxf((float)cnt[g], 1.f);
  float var = vsum[(size_t)g * HD + j] / c;
  y[t] = x[t] * rsqrtf(var + 1e-5f) * w[j] + b[j];
}

__global__ void k_alpha(const float* __restrict__ q, const int* __restrict__ dst,
                        const float* __restrict__ key, float* __restrict__ out, int E) {
  int t = blockIdx.x * blockDim.x + threadIdx.x;
  if (t >= E * HD) return;
  int e = t >> 7, j = t & 127;
  out[t] = q[(size_t)dst[e] * HD + j] * key[t] * 0.08838834764831845f;  // 1/sqrt(128)
}

__global__ void k_mul_pack(const float* __restrict__ a, const float* __restrict__ b,
                           _Float16* __restrict__ o, int n) {
  int t = blockIdx.x * blockDim.x + threadIdx.x;
  if (t < n) o[t] = (_Float16)(a[t] * b[t]);
}

__global__ void k_scale_pack(const float* __restrict__ x, float s,
                             _Float16* __restrict__ o, int n) {
  int t = blockIdx.x * blockDim.x + threadIdx.x;
  if (t < n) o[t] = (_Float16)(x[t] * s);
}

__global__ void k_mixpre(const float* __restrict__ g, const float* __restrict__ a,
                         const float* __restrict__ b, float* __restrict__ o, int n) {
  int t = blockIdx.x * blockDim.x + threadIdx.x;
  if (t >= n) return;
  float gv = g[t];
  o[t] = gv * a[t] + (1.f - gv) * b[t];
}

__global__ void k_resid2(const float* __restrict__ xc, const float* __restrict__ xm,
                         const float* __restrict__ mixed, const float* __restrict__ rsc,
                         const float* __restrict__ rsm, float* __restrict__ xcart,
                         float* __restrict__ xmat, int n) {
  int t = blockIdx.x * blockDim.x + threadIdx.x;
  if (t >= n) return;
  float mv = mixed[t];
  xcart[t] = xc[t] + rsc[0] * mv;
  xmat[t]  = xm[t] + rsm[0] * mv;
}

__global__ void k_avg(const float* __restrict__ a, const float* __restrict__ b,
                      float* __restrict__ o, int n) {
  int t = blockIdx.x * blockDim.x + threadIdx.x;
  if (t < n) o[t] = 0.5f * (a[t] + b[t]);
}

__global__ void k_lstm(const float* __restrict__ g1, const float* __restrict__ g2,
                       float* __restrict__ cc, float* __restrict__ hh,
                       _Float16* __restrict__ hh16, int G) {
  int t = blockIdx.x * blockDim.x + threadIdx.x;
  if (t >= G * HD) return;
  int g = t >> 7, j = t & 127;
  const float* a = g1 + (size_t)g * 512;
  const float* b = g2 + (size_t)g * 512;
  float ig = a[j] + b[j];
  float fg = a[HD + j] + b[HD + j];
  float gg = a[2 * HD + j] + b[2 * HD + j];
  float og = a[3 * HD + j] + b[3 * HD + j];
  float c = sigf(fg) * cc[t] + sigf(ig) * tanhf(gg);
  float h = sigf(og) * tanhf(c);
  cc[t] = c; hh[t] = h; hh16[t] = (_Float16)h;
}

__global__ __launch_bounds__(128)
void k_en(const float* __restrict__ xo, const float* __restrict__ hh,
          const int* __restrict__ batch, float* __restrict__ en, int n) {
  int row = blockIdx.x * 4 + (threadIdx.x >> 5);
  if (row >= n) return;
  int lane = threadIdx.x & 31;
  const float* xr = xo + (size_t)row * HD;
  const float* hr = hh + (size_t)batch[row] * HD;
  float s = 0.f;
  for (int j = 0; j < 4; ++j) s += xr[lane + 32 * j] * hr[lane + 32 * j];
  for (int off = 16; off; off >>= 1) s += __shfl_xor(s, off, 32);
  if (lane == 0) en[row] = s;
}

__global__ void k_softmax_seg(const float* __restrict__ en, const int* __restrict__ batch,
                              float* __restrict__ att, int n) {
  __shared__ float red[256];
  int g = blockIdx.x, tid = threadIdx.x;
  float mx = -3.0e38f;
  for (int i = tid; i < n; i += 256) if (batch[i] == g) mx = fmaxf(mx, en[i]);
  red[tid] = mx; __syncthreads();
  for (int s = 128; s; s >>= 1) { if (tid < s) red[tid] = fmaxf(red[tid], red[tid + s]); __syncthreads(); }
  mx = red[0]; __syncthreads();
  float sum = 0.f;
  for (int i = tid; i < n; i += 256) if (batch[i] == g) sum += expf(en[i] - mx);
  red[tid] = sum; __syncthreads();
  for (int s = 128; s; s >>= 1) { if (tid < s) red[tid] += red[tid + s]; __syncthreads(); }
  float inv = red[0] > 0.f ? 1.f / red[0] : 0.f;
  __syncthreads();
  for (int i = tid; i < n; i += 256) if (batch[i] == g) att[i] = expf(en[i] - mx) * inv;
}

__global__ void k_wsegsum(const float* __restrict__ att, const float* __restrict__ xo,
                          const int* __restrict__ batch, float* __restrict__ r, int n) {
  int t = blockIdx.x * blockDim.x + threadIdx.x;
  if (t >= n * HD) return;
  int i = t >> 7, j = t & 127;
  atomicAdd(&r[(size_t)batch[i] * HD + j], att[i] * xo[t]);
}

__global__ void k_qstar(const float* __restrict__ hh, const float* __restrict__ r,
                        _Float16* __restrict__ qs16, int G) {
  int t = blockIdx.x * blockDim.x + threadIdx.x;
  if (t >= G * HD) return;
  int g = t >> 7, j = t & 127;
  qs16[(size_t)g * 256 + j]      = (_Float16)hh[t];
  qs16[(size_t)g * 256 + HD + j] = (_Float16)r[t];
}

__global__ __launch_bounds__(128)
void k_fc2(const float* __restrict__ t, const float* __restrict__ w,
           const float* __restrict__ b, float* __restrict__ out, int G) {
  int row = blockIdx.x * 4 + (threadIdx.x >> 5);
  if (row >= G) return;
  int lane = threadIdx.x & 31;
  const float* tr = t + (size_t)row * HD;
  float s = 0.f;
  for (int j = 0; j < 4; ++j) s += tr[lane + 32 * j] * w[lane + 32 * j];
  for (int off = 16; off; off >>= 1) s += __shfl_xor(s, off, 32);
  if (lane == 0) out[row] = s + b[0];
}

// ---------------------------------------------------------------------------
// Param indexing: jax pytree (sorted dict keys) appended after 5 data inputs
// ---------------------------------------------------------------------------
enum {
  P0 = 5,
  P_ATOM_EMB = P0 + 0,
  P_ENC_GATE_B = P0 + 1, P_ENC_GATE_W = P0 + 2,
  P_ENC_L1_B = P0 + 3, P_ENC_L1_W = P0 + 4,
  P_ENC_L2_B = P0 + 5, P_ENC_L2_W = P0 + 6,
  P_ENC_LN_B = P0 + 7, P_ENC_LN_W = P0 + 8,
  P_FC1_B = P0 + 9, P_FC1_W = P0 + 10,
  P_FC2_B = P0 + 11, P_FC2_W = P0 + 12,
  P_LAYERS = P0 + 13,
  L_STRIDE = 54,
  L_AGGR_L1_B = 0, L_AGGR_L1_W = 1, L_AGGR_L2_B = 2, L_AGGR_L2_W = 3,
  L_BNE_B = 4, L_BNE_W = 5, L_BNX_B = 6, L_BNX_W = 7,
  L_GATE_L1_B = 8, L_GATE_L1_W = 9, L_GATE_L2_B = 10, L_GATE_L2_W = 11,
  L_FFN_L1_B = 12, L_FFN_L1_W = 13, L_FFN_L2_B = 14, L_FFN_L2_W = 15,
  L_GNC_B = 16, L_GNC_MS = 17, L_GNC_W = 18,
  L_GNM_B = 19, L_GNM_MS = 20, L_GNM_W = 21,
  L_LNPC_B = 22, L_LNPC_W = 23, L_LNPM_B = 24, L_LNPM_W = 25,
  L_E_B = 26, L_E_W = 27, L_K_B = 28, L_K_W = 29,
  L_KEY_L1_B = 30, L_KEY_L1_W = 31, L_KEY_L2_B = 32, L_KEY_L2_W = 33,
  L_LNA_B = 34, L_LNA_W = 35, L_LNM_B = 36, L_LNM_W = 37,
  L_MSGLIN_B = 38, L_MSGLIN_W = 39,
  L_MSG_L1_B = 40, L_MSG_L1_W = 41, L_MSG_L2_B = 42, L_MSG_L2_W = 43,
  L_Q_B = 44, L_Q_W = 45, L_V_B = 46, L_V_W = 47,
  L_MIXG_B = 48, L_MIXG_W = 49, L_MIXLN_B = 50, L_MIXLN_W = 51,
  L_RSC = 52, L_RSM = 53,
  P_LSTM_BHH = P_LAYERS + 5 * L_STRIDE + 0,
  P_LSTM_BIH = P_LSTM_BHH + 1,
  P_LSTM_WHH = P_LSTM_BHH + 2,
  P_LSTM_WIH = P_LSTM_BHH + 3,
  P_MEG_B = P_LSTM_BHH + 4, P_MEG_W = P_LSTM_BHH + 5,
  P_RBF1_B = P_LSTM_BHH + 6, P_RBF1_W = P_LSTM_BHH + 7,
  P_RBF2_B = P_LSTM_BHH + 8, P_RBF2_W = P_LSTM_BHH + 9,
};

#define G1(n) dim3((unsigned)(((n) + 255) / 256))

extern "C" void kernel_launch(void* const* d_in, const int* in_sizes, int n_in,
                              void* d_out, int out_size, void* d_ws, size_t ws_size,
                              hipStream_t stream) {
  (void)in_sizes; (void)n_in; (void)out_size; (void)ws_size;

  const int*   an    = (const int*)d_in[0];
  const float* meg   = (const float*)d_in[1];
  const int*   ei    = (const int*)d_in[2];
  const float* evec  = (const float*)d_in[3];
  const int*   batch = (const int*)d_in[4];
  const int*   srcI  = ei;        // edge_index[0] = src
  const int*   dstI  = ei + NE;   // edge_index[1] = dst
  float* outp = (float*)d_out;

  auto PF = [&](int i) -> const float* { return (const float*)d_in[i]; };

  // ---- deterministic bump allocator over d_ws ----
  char* base = (char*)d_ws;
  size_t off = 0;
  auto alloc = [&](size_t bytes) -> void* {
    off = (off + 255) & ~(size_t)255;
    void* p = base + off;
    off += bytes;
    return p;
  };
  auto fa = [&](size_t n) -> float*    { return (float*)alloc(n * 4); };
  auto ha = [&](size_t n) -> _Float16* { return (_Float16*)alloc(n * 2); };

  // node f32 (padded rows so GEMM stores are unguarded)
  float *embB = fa((size_t)NNP*HD), *mgB = fa((size_t)NNP*HD), *t1 = fa((size_t)NNP*HD);
  float *xcB = fa((size_t)NNP*HD), *xmB = fa((size_t)NNP*HD);
  float *aB = fa((size_t)NNP*HD), *bB = fa((size_t)NNP*HD), *gmB = fa((size_t)NNP*HD);
  float *agg = fa((size_t)NNP*HD);
  float *kH = fa((size_t)NNP*HD), *vH = fa((size_t)NNP*HD), *qH = fa((size_t)NNP*HD);
  float *xOut = fa((size_t)NNP*HD), *xCart = fa((size_t)NNP*HD), *xMat = fa((size_t)NNP*HD);
  // node f16
  _Float16 *fused16 = ha((size_t)NNP*HD), *t2h = ha((size_t)NNP*HD);
  _Float16 *xMat16 = ha((size_t)NNP*HD), *n256h = ha((size_t)NNP*256);
  _Float16 *meg16 = ha((size_t)NNP*32);
  // edge f32 (NE is a 64-multiple)
  float *e0 = fa((size_t)NE*HD), *eCart = fa((size_t)NE*HD);
  float *eNew = fa((size_t)NE*HD), *eT1 = fa((size_t)NE*HD), *eT2 = fa((size_t)NE*HD);
  float *eeH = fa((size_t)NE*HD);
  // edge f16
  _Float16 *e016 = ha((size_t)NE*HD), *eH = ha((size_t)NE*HD), *eH2 = ha((size_t)NE*HD);
  _Float16 *catE = ha((size_t)NE*384);
  // graph / readout (padded where GEMM-written or GEMM-read)
  float *gsum = fa((size_t)NG*HD), *gvsum = fa((size_t)NG*HD), *rG = fa((size_t)NG*HD);
  float *hh = fa((size_t)NG*HD), *cc = fa((size_t)NG*HD), *fcT = fa((size_t)NGP*HD);
  float *gates1 = fa((size_t)NGP*512), *gates2 = fa((size_t)NGP*512);
  float *enB = fa((size_t)NN), *attB = fa((size_t)NN);
  _Float16 *qs16 = ha((size_t)NGP*256), *hh16 = ha((size_t)NGP*HD);
  int *cnt = (int*)alloc((size_t)NG * 4);

  // ---- helpers ----
  auto gemm = [&](const _Float16* A, const _Float16* Bt, const float* bias,
                  float* C, _Float16* C16, int M, int N, int K, int act) {
    k_gemm_wmma<<<dim3((unsigned)((M + 63) / 64), (unsigned)(N / 64)), 128, 0, stream>>>(
        A, Bt, bias, C, C16, N, K, act);
  };
  auto prepW = [&](int idx, int K, int N) -> _Float16* {
    _Float16* Wt = ha((size_t)K * N);
    k_transpose_cvt<<<G1(K * N), 256, 0, stream>>>(PF(idx), Wt, K, K, N);
    return Wt;
  };
  auto ln = [&](const float* x, const float* w, const float* b, float* y,
                _Float16* y16, int rows, int act) {
    k_layernorm128<<<dim3((unsigned)((rows + 3) / 4)), 128, 0, stream>>>(x, w, b, y, y16, rows, act);
  };
  auto gnorm = [&](float* x, const float* gb, const float* gms, const float* gw) {
    hipMemsetAsync(gsum, 0, (size_t)NG * HD * 4, stream);
    hipMemsetAsync(gvsum, 0, (size_t)NG * HD * 4, stream);
    k_seg_accum<<<G1(NN * HD), 256, 0, stream>>>(x, batch, gsum, NN);
    k_gn1<<<G1(NN * HD), 256, 0, stream>>>(x, gsum, cnt, gms, batch, x, gvsum, NN);
    k_gn2<<<G1(NN * HD), 256, 0, stream>>>(x, gvsum, cnt, batch, gw, gb, x, NN);
  };

  // ---- per-graph node counts (batch is fixed input) ----
  hipMemsetAsync(cnt, 0, (size_t)NG * 4, stream);
  k_count<<<G1(NN), 256, 0, stream>>>(batch, cnt, NN);

  // ---- convert global (non-layer) weights to f16 transposed ----
  _Float16* WtMeg = ha((size_t)32 * HD);   // K padded 16 -> 32
  k_transpose_cvt<<<G1(32 * HD), 256, 0, stream>>>(PF(P_MEG_W), WtMeg, 16, 32, HD);
  _Float16* WtEncGate = prepW(P_ENC_GATE_W, 256, HD);
  _Float16* WtEnc1    = prepW(P_ENC_L1_W, HD, HD);
  _Float16* WtEnc2    = prepW(P_ENC_L2_W, HD, HD);
  _Float16* WtRbf1    = prepW(P_RBF1_W, HD, HD);
  _Float16* WtRbf2    = prepW(P_RBF2_W, HD, HD);
  _Float16* WtWih     = prepW(P_LSTM_WIH, 256, 512);
  _Float16* WtWhh     = prepW(P_LSTM_WHH, HD, 512);
  _Float16* WtFc1     = prepW(P_FC1_W, 256, HD);

  // ---- AtomEncoder ----
  k_embed<<<G1(NN * HD), 256, 0, stream>>>(an, PF(P_ATOM_EMB), embB, NN);
  k_pack_meg<<<G1(NN * 32), 256, 0, stream>>>(meg, meg16, NN);
  gemm(meg16, WtMeg, PF(P_MEG_B), mgB, nullptr, NN, HD, 32, 0);
  k_cat2<<<G1(NN * 256), 256, 0, stream>>>(embB, mgB, n256h, NN);
  gemm(n256h, WtEncGate, PF(P_ENC_GATE_B), gmB, nullptr, NN, HD, 256, 2);
  k_fuse<<<G1(NN * HD), 256, 0, stream>>>(gmB, embB, mgB, fused16, NN * HD);
  gemm(fused16, WtEnc1, PF(P_ENC_L1_B), t1, nullptr, NN, HD, HD, 1);
  ln(t1, PF(P_ENC_LN_W), PF(P_ENC_LN_B), t1, t2h, NN, 0);
  gemm(t2h, WtEnc2, PF(P_ENC_L2_B), xCart, nullptr, NN, HD, HD, 0);
  hipMemcpyAsync(xMat, xCart, (size_t)NNP * HD * 4, hipMemcpyDeviceToDevice, stream);

  // ---- RBF edge embedding ----
  k_rbf<<<G1(NE * HD), 256, 0, stream>>>(evec, eH, NE);
  gemm(eH, WtRbf1, PF(P_RBF1_B), nullptr, eH2, NE, HD, HD, 1);
  gemm(eH2, WtRbf2, PF(P_RBF2_B), e0, nullptr, NE, HD, HD, 0);
  k_pack<<<G1(NE * HD), 256, 0, stream>>>(e0, e016, NE * HD);
  hipMemcpyAsync(eCart, e0, (size_t)NE * HD * 4, hipMemcpyDeviceToDevice, stream);

  // ---- layers ----
  for (int L = 0; L < 5; ++L) {
    const int LB = P_LAYERS + L * L_STRIDE;
    _Float16* WtGate1  = prepW(LB + L_GATE_L1_W, 384, HD);
    _Float16* WtGate2  = prepW(LB + L_GATE_L2_W, HD, HD);
    _Float16* WtAggr1  = prepW(LB + L_AGGR_L1_W, 384, HD);
    _Float16* WtAggr2  = prepW(LB + L_AGGR_L2_W, HD, HD);
    _Float16* WtQ      = prepW(LB + L_Q_W, HD, 512);
    _Float16* WtK      = prepW(LB + L_K_W, HD, 512);
    _Float16* WtV      = prepW(LB + L_V_W, HD, 512);
    _Float16* WtE      = prepW(LB + L_E_W, HD, 512);
    _Float16* WtKey1   = prepW(LB + L_KEY_L1_W, 384, HD);
    _Float16* WtKey2   = prepW(LB + L_KEY_L2_W, HD, HD);
    _Float16* WtMsg1   = prepW(LB + L_MSG_L1_W, 384, HD);
    _Float16* WtMsg2   = prepW(LB + L_MSG_L2_W, HD, HD);
    _Float16* WtMsgLin = prepW(LB + L_MSGLIN_W, HD, HD);
    _Float16* WtFfn1   = prepW(LB + L_FFN_L1_W, HD, 256);
    _Float16* WtFfn2   = prepW(LB + L_FFN_L2_W, 256, HD);
    _Float16* WtMix    = prepW(LB + L_MIXG_W, 256, HD);

    // --- CartNet branch ---
    k_gather_cat3<<<G1(NE * HD), 256, 0, stream>>>(xCart, eCart, dstI, srcI, catE, NE);
    gemm(catE, WtGate1, PF(LB + L_GATE_L1_B), nullptr, eH, NE, HD, 384, 1);
    gemm(eH, WtGate2, PF(LB + L_GATE_L2_B), eNew, nullptr, NE, HD, HD, 0);
    gemm(catE, WtAggr1, PF(LB + L_AGGR_L1_B), nullptr, eH, NE, HD, 384, 1);
    gemm(eH, WtAggr2, PF(LB + L_AGGR_L2_B), eT1, nullptr, NE, HD, HD, 0);
    k_gate_mul<<<G1(NE * HD), 256, 0, stream>>>(eNew, PF(LB + L_BNE_W), PF(LB + L_BNE_B), eT1, eT2, NE);
    hipMemsetAsync(agg, 0, (size_t)NNP * HD * 4, stream);
    k_scatter_add<<<G1(NE * HD), 256, 0, stream>>>(eT2, dstI, agg, NE);
    k_silu_affine_add<<<G1(NN * HD), 256, 0, stream>>>(agg, PF(LB + L_BNX_W), PF(LB + L_BNX_B), xCart, xcB, NN);
    k_add<<<G1(NE * HD), 256, 0, stream>>>(eCart, eNew, eCart, NE * HD);
    gnorm(xcB, PF(LB + L_GNC_B), PF(LB + L_GNC_MS), PF(LB + L_GNC_W));

    // --- Matformer branch (per-head) ---
    k_pack<<<G1(NN * HD), 256, 0, stream>>>(xMat, xMat16, NN * HD);
    hipMemsetAsync(agg, 0, (size_t)NNP * HD * 4, stream);  // msum across heads
    for (int h = 0; h < 4; ++h) {
      const int wo = h * HD * HD, bo = h * HD;
      gemm(xMat16, WtK + wo, PF(LB + L_K_B) + bo, kH, nullptr, NN, HD, HD, 0);
      gemm(xMat16, WtV + wo, PF(LB + L_V_B) + bo, vH, nullptr, NN, HD, HD, 0);
      gemm(xMat16, WtQ + wo, PF(LB + L_Q_B) + bo, qH, nullptr, NN, HD, HD, 0);
      gemm(e016, WtE + wo, PF(LB + L_E_B) + bo, eeH, nullptr, NE, HD, HD, 0);
      k_gather_cat3<<<G1(NE * HD), 256, 0, stream>>>(kH, eeH, dstI, srcI, catE, NE);
      gemm(catE, WtKey1, PF(LB + L_KEY_L1_B), nullptr, eH, NE, HD, 384, 1);
      gemm(eH, WtKey2, PF(LB + L_KEY_L2_B), eT1, nullptr, NE, HD, HD, 0);       // key
      k_alpha<<<G1(NE * HD), 256, 0, stream>>>(qH, dstI, eT1, eT2, NE);         // alpha
      ln(eT2, PF(LB + L_LNA_W), PF(LB + L_LNA_B), eT2, nullptr, NE, 1);         // sigmoid(LN)
      k_gather_cat3<<<G1(NE * HD), 256, 0, stream>>>(vH, eeH, dstI, srcI, catE, NE);
      gemm(catE, WtMsg1, PF(LB + L_MSG_L1_B), nullptr, eH, NE, HD, 384, 1);
      gemm(eH, WtMsg2, PF(LB + L_MSG_L2_B), eT1, nullptr, NE, HD, HD, 0);       // m
      k_mul_pack<<<G1(NE * HD), 256, 0, stream>>>(eT1, eT2, eH, NE * HD);       // m*sa -> f16
      gemm(eH, WtMsgLin, PF(LB + L_MSGLIN_B), eT1, nullptr, NE, HD, HD, 0);
      ln(eT1, PF(LB + L_LNM_W), PF(LB + L_LNM_B), eT1, nullptr, NE, 0);
      k_scatter_add<<<G1(NE * HD), 256, 0, stream>>>(eT1, dstI, agg, NE);
    }
    k_scale_pack<<<G1(NN * HD), 256, 0, stream>>>(agg, 0.25f, xMat16, NN * HD); // mean over heads
    gemm(xMat16, WtFfn1, PF(LB + L_FFN_L1_B), nullptr, n256h, NN, 256, HD, 1);
    gemm(n256h, WtFfn2, PF(LB + L_FFN_L2_B), xmB, nullptr, NN, HD, 256, 0);
    gnorm(xmB, PF(LB + L_GNM_B), PF(LB + L_GNM_MS), PF(LB + L_GNM_W));

    // --- ResidualGateMixer ---
    ln(xcB, PF(LB + L_LNPC_W), PF(LB + L_LNPC_B), aB, nullptr, NN, 0);
    ln(xmB, PF(LB + L_LNPM_W), PF(LB + L_LNPM_B), bB, nullptr, NN, 0);
    k_cat2<<<G1(NN * 256), 256, 0, stream>>>(aB, bB, n256h, NN);
    gemm(n256h, WtMix, PF(LB + L_MIXG_B), gmB, nullptr, NN, HD, 256, 2);
    k_mixpre<<<G1(NN * HD), 256, 0, stream>>>(gmB, aB, bB, t1, NN * HD);
    ln(t1, PF(LB + L_MIXLN_W), PF(LB + L_MIXLN_B), t1, nullptr, NN, 0);
    k_resid2<<<G1(NN * HD), 256, 0, stream>>>(xcB, xmB, t1, PF(LB + L_RSC), PF(LB + L_RSM),
                                              xCart, xMat, NN * HD);
  }

  // ---- Set2Set readout ----
  k_avg<<<G1(NN * HD), 256, 0, stream>>>(xCart, xMat, xOut, NN * HD);
  hipMemsetAsync(qs16, 0, (size_t)NGP * 256 * 2, stream);
  hipMemsetAsync(hh, 0, (size_t)NG * HD * 4, stream);
  hipMemsetAsync(hh16, 0, (size_t)NGP * HD * 2, stream);
  hipMemsetAsync(cc, 0, (size_t)NG * HD * 4, stream);
  for (int s = 0; s < 3; ++s) {
    gemm(qs16, WtWih, PF(P_LSTM_BIH), gates1, nullptr, NG, 512, 256, 0);
    gemm(hh16, WtWhh, PF(P_LSTM_BHH), gates2, nullptr, NG, 512, HD, 0);
    k_lstm<<<G1(NG * HD), 256, 0, stream>>>(gates1, gates2, cc, hh, hh16, NG);
    k_en<<<dim3((unsigned)((NN + 3) / 4)), 128, 0, stream>>>(xOut, hh, batch, enB, NN);
    k_softmax_seg<<<dim3(NG), 256, 0, stream>>>(enB, batch, attB, NN);
    hipMemsetAsync(rG, 0, (size_t)NG * HD * 4, stream);
    k_wsegsum<<<G1(NN * HD), 256, 0, stream>>>(attB, xOut, batch, rG, NN);
    k_qstar<<<G1(NG * HD), 256, 0, stream>>>(hh, rG, qs16, NG);
  }
  gemm(qs16, WtFc1, PF(P_FC1_B), fcT, nullptr, NG, HD, 256, 1);
  k_fc2<<<dim3((unsigned)((NG + 3) / 4)), 128, 0, stream>>>(fcT, PF(P_FC2_W), PF(P_FC2_B), outp, NG);
}
